// RiskAnalyzerGCN_35682588295886
// MI455X (gfx1250) — matrix operand order repaired and checked by reading, not compiled, not run on hardware
//
#include <hip/hip_runtime.h>
#include <hip/hip_fp16.h>

typedef __attribute__((ext_vector_type(16))) _Float16 v16h;
typedef __attribute__((ext_vector_type(8)))  float    v8f;

#define GCN_NODES 50000
#define GCN_EDGES 800000
#define GCN_IN    768
#define GCN_HID   128
#define GCN_CLS   6

static __device__ __forceinline__ int clampi(int v, int himax) {
  return v < 0 ? 0 : (v > himax ? himax : v);
}

// ---------------- degree / normalization ----------------
__global__ void gcn_init_deg(float* deg, int n) {
  int i = blockIdx.x * blockDim.x + threadIdx.x;
  if (i < n) deg[i] = 1.0f;  // self-loop contributes 1
}

__global__ void gcn_degree(const int* __restrict__ dst, float* __restrict__ deg,
                           int nE, int nN) {
  int e = blockIdx.x * blockDim.x + threadIdx.x;
  if (e < nE) {
    int d = clampi(dst[e], nN - 1);
    unsafeAtomicAdd(&deg[d], 1.0f);
  }
}

__global__ void gcn_dinv(float* deg, int n) {
  int i = blockIdx.x * blockDim.x + threadIdx.x;
  if (i < n) deg[i] = rsqrtf(deg[i]);  // deg >= 1 always
}

// ---------------- weight packing into WMMA B-fragment layout ----------------
// B fragment (16-bit, 32x16, wave32): lane l holds column n = l&15,
// h = l>>4; element e -> K = 8h + (e&7) + (e>=8 ? 16 : 0) within the 32-K step.
// Packed index: ((nt*KS + ks)*32 + lane)*16 + e   (32B contiguous per lane)
__global__ void gcn_packW(const float* __restrict__ W, _Float16* __restrict__ hi,
                          _Float16* __restrict__ lo, int KS) {
  int idx = blockIdx.x * blockDim.x + threadIdx.x;
  int total = 8 * KS * 512;
  if (idx >= total) return;
  int e    = idx & 15;
  int lane = (idx >> 4) & 31;
  int ks   = (idx >> 9) % KS;
  int nt   = idx / (KS * 512);
  int n = nt * 16 + (lane & 15);
  int k = ks * 32 + ((lane >> 4) << 3) + (e & 7) + ((e >> 3) << 4);
  float v = W[k * GCN_HID + n];
  _Float16 h = (_Float16)v;
  hi[idx] = h;
  lo[idx] = (_Float16)(v - (float)h);
}

// ---------------- WMMA GEMM: Out[M x 128] = X[M x K] * W[K x 128] ----------------
// One wave computes a 16 x 128 output strip (8 n-tiles), K-loop in steps of 32.
// Split-f16 compensation: 3 WMMAs per tile -> near-f32 accuracy, f32 accumulate.
__global__ __launch_bounds__(256) void gcn_gemm_wmma(
    const float* __restrict__ X, const _Float16* __restrict__ Bhi,
    const _Float16* __restrict__ Blo, float* __restrict__ Out,
    int Mtiles, int K, int KS) {
  int wave = threadIdx.x >> 5;
  int lane = threadIdx.x & 31;
  int mtile = blockIdx.x * 8 + wave;
  if (mtile >= Mtiles) return;

  // A fragment (16x32 f16): lane l -> row m = l&15;
  // lanes 0-15: elems 0..7 = K kb..kb+7, elems 8..15 = K kb+16..kb+23 (kb0=0)
  // lanes 16-31: same with kb0 = 8
  int m   = mtile * 16 + (lane & 15);
  int kb0 = (lane >> 4) * 8;
  const float* xrow = X + (long)m * K;

  v8f zero = {};
  v8f acc[8];
#pragma unroll
  for (int nt = 0; nt < 8; ++nt) acc[nt] = zero;

  for (int ks = 0; ks < KS; ++ks) {
    int kb = ks * 32 + kb0;
    float4 f0 = *(const float4*)(xrow + kb);
    float4 f1 = *(const float4*)(xrow + kb + 4);
    float4 f2 = *(const float4*)(xrow + kb + 16);
    float4 f3 = *(const float4*)(xrow + kb + 20);
    if (ks + 1 < KS) __builtin_prefetch(xrow + kb + 32, 0, 1);

    float xs[16] = {f0.x, f0.y, f0.z, f0.w, f1.x, f1.y, f1.z, f1.w,
                    f2.x, f2.y, f2.z, f2.w, f3.x, f3.y, f3.z, f3.w};
    v16h ahi, alo;
#pragma unroll
    for (int j = 0; j < 16; ++j) {
      _Float16 h = (_Float16)xs[j];
      ahi[j] = h;
      alo[j] = (_Float16)(xs[j] - (float)h);
    }

#pragma unroll
    for (int nt = 0; nt < 8; ++nt) {
      long boff = (((long)nt * KS + ks) * 32 + lane) * 16;
      v16h bh = *(const v16h*)(Bhi + boff);
      v16h bl = *(const v16h*)(Blo + boff);
      acc[nt] = __builtin_amdgcn_wmma_f32_16x16x32_f16(
          false, ahi, false, bh, (short)0, acc[nt], false, false);
      acc[nt] = __builtin_amdgcn_wmma_f32_16x16x32_f16(
          false, alo, false, bh, (short)0, acc[nt], false, false);
      acc[nt] = __builtin_amdgcn_wmma_f32_16x16x32_f16(
          false, ahi, false, bl, (short)0, acc[nt], false, false);
    }
  }

  // C/D layout: VGPR r: lanes 0-15 -> M=r, lanes 16-31 -> M=r+8; N = lane&15
  int rbase = mtile * 16 + ((lane >> 4) << 3);
  int cbase = (lane & 15);
#pragma unroll
  for (int nt = 0; nt < 8; ++nt) {
#pragma unroll
    for (int r = 0; r < 8; ++r) {
      Out[(long)(rbase + r) * GCN_HID + nt * 16 + cbase] = acc[nt][r];
    }
  }
}

// ---------------- zero scratch ----------------
__global__ void gcn_zero(float* p, long n) {
  long i = (long)blockIdx.x * blockDim.x + threadIdx.x;
  if (i < n) p[i] = 0.0f;
}

// ---------------- edge scatter-add: one wave per edge, 4 channels/lane ---------
__global__ __launch_bounds__(256) void gcn_aggregate(
    const float* __restrict__ H, const int* __restrict__ src,
    const int* __restrict__ dst, const float* __restrict__ dinv,
    float* __restrict__ Agg, int nE, int nN) {
  int wave = threadIdx.x >> 5;
  int lane = threadIdx.x & 31;
  long e = (long)blockIdx.x * 8 + wave;
  if (e >= nE) return;
  int s = clampi(src[e], nN - 1);
  int d = clampi(dst[e], nN - 1);
  float nrm = dinv[s] * dinv[d];
  float4 v = *(const float4*)(H + (long)s * GCN_HID + lane * 4);
  float* o = Agg + (long)d * GCN_HID + lane * 4;
  unsafeAtomicAdd(o + 0, v.x * nrm);
  unsafeAtomicAdd(o + 1, v.y * nrm);
  unsafeAtomicAdd(o + 2, v.z * nrm);
  unsafeAtomicAdd(o + 3, v.w * nrm);
}

// -------- fused self-loop + bias + ReLU: Agg = relu(Agg + dinv^2*Hpre + b) ----
__global__ void gcn_selfloop_bias_relu(float* __restrict__ Agg,
                                       const float* __restrict__ Hpre,
                                       const float* __restrict__ dinv,
                                       const float* __restrict__ bias, long n) {
  long idx = (long)blockIdx.x * blockDim.x + threadIdx.x;
  if (idx >= n) return;
  int i = (int)(idx >> 7);
  int c = (int)(idx & 127);
  float di = dinv[i];
  float v = Agg[idx] + di * di * Hpre[idx] + bias[c];
  Agg[idx] = v > 0.0f ? v : 0.0f;
}

// ---------------- heads: two 128->6 linears + log_softmax, wave per node ------
__global__ __launch_bounds__(256) void gcn_heads(
    const float* __restrict__ H, const float* __restrict__ Wi,
    const float* __restrict__ bi, const float* __restrict__ Wl,
    const float* __restrict__ bl, float* __restrict__ out, int nN) {
  int wave = threadIdx.x >> 5;
  int lane = threadIdx.x & 31;
  int node = blockIdx.x * 8 + wave;
  if (node >= nN) return;
  float4 h4 = *(const float4*)(H + (long)node * GCN_HID + lane * 4);
  float ha[4] = {h4.x, h4.y, h4.z, h4.w};
  float pi[GCN_CLS], pl[GCN_CLS];
#pragma unroll
  for (int c = 0; c < GCN_CLS; ++c) { pi[c] = 0.0f; pl[c] = 0.0f; }
#pragma unroll
  for (int j = 0; j < 4; ++j) {
    int k = lane * 4 + j;
    float hv = ha[j];
#pragma unroll
    for (int c = 0; c < GCN_CLS; ++c) {
      pi[c] += hv * Wi[k * GCN_CLS + c];
      pl[c] += hv * Wl[k * GCN_CLS + c];
    }
  }
#pragma unroll
  for (int c = 0; c < GCN_CLS; ++c) {
    for (int off = 16; off > 0; off >>= 1) {
      pi[c] += __shfl_xor(pi[c], off, 32);
      pl[c] += __shfl_xor(pl[c], off, 32);
    }
  }
  if (lane == 0) {
    float li[GCN_CLS], ll[GCN_CLS];
#pragma unroll
    for (int c = 0; c < GCN_CLS; ++c) {
      li[c] = pi[c] + bi[c];
      ll[c] = pl[c] + bl[c];
    }
    float mi = li[0], ml = ll[0];
#pragma unroll
    for (int c = 1; c < GCN_CLS; ++c) {
      mi = fmaxf(mi, li[c]);
      ml = fmaxf(ml, ll[c]);
    }
    float si = 0.0f, sl = 0.0f;
#pragma unroll
    for (int c = 0; c < GCN_CLS; ++c) {
      si += expf(li[c] - mi);
      sl += expf(ll[c] - ml);
    }
    float lsei = mi + logf(si);
    float lsel = ml + logf(sl);
    float* oi = out + (long)node * GCN_CLS;
    float* ol = out + (long)nN * GCN_CLS + (long)node * GCN_CLS;
#pragma unroll
    for (int c = 0; c < GCN_CLS; ++c) {
      oi[c] = li[c] - lsei;
      ol[c] = ll[c] - lsel;
    }
  }
}

extern "C" void kernel_launch(void* const* d_in, const int* in_sizes, int n_in,
                              void* d_out, int out_size, void* d_ws, size_t ws_size,
                              hipStream_t stream) {
  const float* x  = (const float*)d_in[0];
  const int*   ei = (const int*)d_in[1];   // JAX default x64-disabled => int32
  const float* W1 = (const float*)d_in[2];
  const float* b1 = (const float*)d_in[3];
  const float* W2 = (const float*)d_in[4];
  const float* b2 = (const float*)d_in[5];
  const float* Wi = (const float*)d_in[6];
  const float* bi = (const float*)d_in[7];
  const float* Wl = (const float*)d_in[8];
  const float* bl = (const float*)d_in[9];
  const int* src = ei;
  const int* dst = ei + GCN_EDGES;
  float* out = (float*)d_out;

  // ---- workspace carve-out (256B-aligned slabs) ----
  char* ws = (char*)d_ws;
  size_t off = 0;
  auto carve = [&](size_t bytes) -> void* {
    void* p = (void*)(ws + off);
    off += (bytes + 255) & ~(size_t)255;
    return p;
  };
  const size_t HBYTES = (size_t)GCN_NODES * GCN_HID * sizeof(float);
  float*    buf0 = (float*)carve(HBYTES);            // pre-agg h (layer in flight)
  float*    buf1 = (float*)carve(HBYTES);            // agg / post-relu layer 1
  float*    buf2 = (float*)carve(HBYTES);            // agg / post-relu layer 2
  float*    dinv = (float*)carve((size_t)GCN_NODES * sizeof(float));
  _Float16* w1hi = (_Float16*)carve((size_t)8 * 24 * 512 * sizeof(_Float16));
  _Float16* w1lo = (_Float16*)carve((size_t)8 * 24 * 512 * sizeof(_Float16));
  _Float16* w2hi = (_Float16*)carve((size_t)8 * 4 * 512 * sizeof(_Float16));
  _Float16* w2lo = (_Float16*)carve((size_t)8 * 4 * 512 * sizeof(_Float16));
  (void)ws_size; (void)in_sizes; (void)n_in; (void)out_size;

  const int Mtiles = GCN_NODES / 16;                 // 3125 exactly
  const long HN = (long)GCN_NODES * GCN_HID;

  // ---- degree + symmetric normalization ----
  gcn_init_deg<<<(GCN_NODES + 255) / 256, 256, 0, stream>>>(dinv, GCN_NODES);
  gcn_degree<<<(GCN_EDGES + 255) / 256, 256, 0, stream>>>(dst, dinv, GCN_EDGES, GCN_NODES);
  gcn_dinv<<<(GCN_NODES + 255) / 256, 256, 0, stream>>>(dinv, GCN_NODES);

  // ---- pack weights into WMMA B-fragment hi/lo halves ----
  gcn_packW<<<(8 * 24 * 512 + 255) / 256, 256, 0, stream>>>(W1, w1hi, w1lo, 24);
  gcn_packW<<<(8 * 4 * 512 + 255) / 256, 256, 0, stream>>>(W2, w2hi, w2lo, 4);

  // ---- layer 1: h1pre = x @ W1 ----
  gcn_gemm_wmma<<<(Mtiles + 7) / 8, 256, 0, stream>>>(x, w1hi, w1lo, buf0,
                                                      Mtiles, GCN_IN, 24);
  gcn_zero<<<(int)((HN + 255) / 256), 256, 0, stream>>>(buf1, HN);
  gcn_aggregate<<<(GCN_EDGES + 7) / 8, 256, 0, stream>>>(buf0, src, dst, dinv,
                                                         buf1, GCN_EDGES, GCN_NODES);
  gcn_selfloop_bias_relu<<<(int)((HN + 255) / 256), 256, 0, stream>>>(buf1, buf0,
                                                                      dinv, b1, HN);

  // ---- layer 2: h2pre = h1 @ W2 ----
  gcn_gemm_wmma<<<(Mtiles + 7) / 8, 256, 0, stream>>>(buf1, w2hi, w2lo, buf0,
                                                      Mtiles, GCN_HID, 4);
  gcn_zero<<<(int)((HN + 255) / 256), 256, 0, stream>>>(buf2, HN);
  gcn_aggregate<<<(GCN_EDGES + 7) / 8, 256, 0, stream>>>(buf0, src, dst, dinv,
                                                         buf2, GCN_EDGES, GCN_NODES);
  gcn_selfloop_bias_relu<<<(int)((HN + 255) / 256), 256, 0, stream>>>(buf2, buf0,
                                                                      dinv, b2, HN);

  // ---- heads + log_softmax ----
  gcn_heads<<<(GCN_NODES + 7) / 8, 256, 0, stream>>>(buf2, Wi, bi, Wl, bl,
                                                     out, GCN_NODES);
}
